// Geoattn_GNN_55044300866310
// MI455X (gfx1250) — matrix operand order repaired and checked by reading, not compiled, not run on hardware
//
#include <hip/hip_runtime.h>

#define NN 20000
#define NE 320000

typedef __attribute__((ext_vector_type(16))) _Float16 v16h;
typedef __attribute__((ext_vector_type(8)))  _Float16 v8h;
typedef __attribute__((ext_vector_type(2)))  _Float16 v2h;
typedef __attribute__((ext_vector_type(8)))  float    v8f;

// ---------------- WMMA helpers (gfx1250 wave32 layouts) ----------------

__device__ __forceinline__ v8f wmma_f16(v16h a, v16h b, v8f c) {
  return __builtin_amdgcn_wmma_f32_16x16x32_f16(false, a, false, b, (short)0, c,
                                                false, false);
}

// A fragment: 16x32 f16 tile from LDS staging (row-major, lda halfwords,
// lda multiple of 32, rows zero-padded). Lane L: M=L&15, K-base=8*(L>>4).
__device__ __forceinline__ v16h load_a_frag(const _Float16* st, int lda, int kc,
                                            int lane) {
  int m = lane & 15;
  int kb = kc * 32 + ((lane >> 4) << 3);
  const _Float16* p = st + m * lda + kb;
  v8h lo = *(const v8h*)(p);
  v8h hi = *(const v8h*)(p + 16);
  v16h a;
#pragma unroll
  for (int j = 0; j < 8; ++j) { a[j] = lo[j]; a[j + 8] = hi[j]; }
  return a;
}

// One-wave GEMM: C(16 x OUT, f32, LDS) = A(16 x K f16 staged) * W^T (+bias).
// Bf = pre-swizzled f16 B fragments: frag (nt*kC+kc) is 512 halfs,
// lane's 16 values at frag*512 + lane*16 (32B aligned, contiguous per lane).
// A fragments hoisted once; all WMMAs issued back-to-back, then stores.
template <int K, int OUT, bool HASB>
__device__ __forceinline__ void wave_gemm(const _Float16* A, int lda,
                                          const _Float16* Bf, const float* bias,
                                          float* C, int ldc, int lane) {
  constexpr int nT = OUT >> 4;
  constexpr int kC = (K + 31) >> 5;
  v16h afr[kC];
#pragma unroll
  for (int kc = 0; kc < kC; ++kc) afr[kc] = load_a_frag(A, lda, kc, lane);
  v8f acc[nT];
#pragma unroll
  for (int nt = 0; nt < nT; ++nt) {
    acc[nt] = {};
#pragma unroll
    for (int kc = 0; kc < kC; ++kc) {
      const v8h* bp = (const v8h*)(Bf + ((nt * kC + kc) << 9) + (lane << 4));
      v8h b0 = bp[0], b1 = bp[1];
      v16h b;
#pragma unroll
      for (int j = 0; j < 8; ++j) { b[j] = b0[j]; b[j + 8] = b1[j]; }
      acc[nt] = wmma_f16(afr[kc], b, acc[nt]);
    }
  }
#pragma unroll
  for (int nt = 0; nt < nT; ++nt) {
    int n = (nt << 4) + (lane & 15);
    float bv = 0.0f;
    if (HASB) bv = bias[n];
#pragma unroll
    for (int v = 0; v < 8; ++v) {
      float x = acc[nt][v];
      if (HASB) x += bv;
      C[(v + ((lane >> 4) << 3)) * ldc + n] = x;
    }
  }
}

// Branch-free staging: 16 x SRCW f32 (row stride SSTRIDE) -> f16 16 x LDA,
// pair-packed writes, unconditional zero-fill of pad columns.
template <int SRCW, int SSTRIDE, int LDA>
__device__ __forceinline__ void stage_f32(_Float16* dst, const float* src,
                                          int lane) {
#pragma unroll
  for (int t = lane; t < 16 * SRCW / 2; t += 32) {
    int m = t / (SRCW / 2), p = t % (SRCW / 2);
    v2h h;
    h[0] = (_Float16)src[m * SSTRIDE + 2 * p];
    h[1] = (_Float16)src[m * SSTRIDE + 2 * p + 1];
    ((v2h*)dst)[m * (LDA / 2) + p] = h;
  }
  if constexpr (LDA > SRCW) {
#pragma unroll
    for (int t = lane; t < 16 * (LDA - SRCW) / 2; t += 32) {
      int m = t / ((LDA - SRCW) / 2), p = t % ((LDA - SRCW) / 2);
      v2h z = {};
      ((v2h*)dst)[m * (LDA / 2) + SRCW / 2 + p] = z;
    }
  }
}

__device__ __forceinline__ void atomicMaxF(float* addr, float val) {
  if (val >= 0.0f) atomicMax((int*)addr, __float_as_int(val));
  else             atomicMin((unsigned int*)addr, __float_as_uint(val));
}

__device__ __forceinline__ float sigmoidf(float x) {
  return 1.0f / (1.0f + expf(-x));
}

// ---------------- per-wave LDS workspace ----------------

struct __align__(16) ShmG {
  _Float16 stA[16 * 96];      // staging, K padded up to 96
  float res[16 * 64];         // GEMM scalar outputs / out_s
  float vi[3 * 16 * 32];      // GVLinear hidden vectors (stride 32)
  float vbuf[3 * 16 * 32];    // general vector buffer (stride 32)
  float vbuf2[3 * 16 * 16];   // out_v (stride 16)
  float vin[3 * 16 * 16];     // vector inputs (stride 16)
  float tmp[16 * 32];
  float tmp2[16 * 32];
  float sc[16 * 8];
  int   idx[32];
};

// Generic GVLinear over a 16-row tile (one wave). outs=64, outv=16.
// vIn layout: vIn[(d*16+m)*VSTRIDE + k], k < INV. Results: out_s -> S->res,
// gated out_v -> S->vbuf2 ([d][m][i], stride 16).
template <int INS, int INV, int DH, int VSTRIDE>
__device__ __forceinline__ void gv_forward(ShmG* S, const float* sIn,
                                           const float* vIn,
                                           const _Float16* BWv1,
                                           const _Float16* BWv2,
                                           const _Float16* BWs,
                                           const _Float16* BWg,
                                           const float* bg, int lane) {
  // vi[d] = Wv1 @ vIn[d]
  for (int d = 0; d < 3; ++d) {
    stage_f32<INV, VSTRIDE, 32>(S->stA, vIn + d * 16 * VSTRIDE, lane);
    __syncthreads();
    wave_gemm<INV, DH, false>(S->stA, 32, BWv1, nullptr, S->vi + d * 512, 32,
                              lane);
    __syncthreads();
  }
  // out_s = Ws @ [vnorm(DH), sIn(INS)]
  constexpr int K = DH + INS;
  constexpr int LDA = (K + 31) & ~31;
#pragma unroll
  for (int t = lane; t < 16 * DH / 2; t += 32) {  // vnorm pairs
    int m = t / (DH / 2), p = t % (DH / 2);
    v2h h;
#pragma unroll
    for (int q = 0; q < 2; ++q) {
      int k = 2 * p + q;
      float x0 = S->vi[m * 32 + k];
      float x1 = S->vi[512 + m * 32 + k];
      float x2 = S->vi[1024 + m * 32 + k];
      h[q] = (_Float16)sqrtf(x0 * x0 + x1 * x1 + x2 * x2);
    }
    ((v2h*)S->stA)[m * (LDA / 2) + p] = h;
  }
#pragma unroll
  for (int t = lane; t < 16 * INS / 2; t += 32) {  // sIn pairs
    int m = t / (INS / 2), p = t % (INS / 2);
    v2h h;
    h[0] = (_Float16)sIn[m * INS + 2 * p];
    h[1] = (_Float16)sIn[m * INS + 2 * p + 1];
    ((v2h*)S->stA)[m * (LDA / 2) + DH / 2 + p] = h;
  }
  if constexpr (LDA > K) {
#pragma unroll
    for (int t = lane; t < 16 * (LDA - K) / 2; t += 32) {  // pad pairs
      int m = t / ((LDA - K) / 2), p = t % ((LDA - K) / 2);
      v2h z = {};
      ((v2h*)S->stA)[m * (LDA / 2) + K / 2 + p] = z;
    }
  }
  __syncthreads();
  wave_gemm<K, 64, false>(S->stA, LDA, BWs, nullptr, S->res, 64, lane);
  __syncthreads();
  // out_v[d] = Wv2 @ vi[d]
  for (int d = 0; d < 3; ++d) {
    stage_f32<DH, 32, 32>(S->stA, S->vi + d * 512, lane);
    __syncthreads();
    wave_gemm<DH, 16, false>(S->stA, 32, BWv2, nullptr, S->vbuf2 + d * 256, 16,
                             lane);
    __syncthreads();
  }
  // gate = sigmoid(Wg @ out_s + bg); out_v *= gate
  stage_f32<64, 64, 64>(S->stA, S->res, lane);
  __syncthreads();
  wave_gemm<64, 16, true>(S->stA, 64, BWg, bg, S->tmp, 16, lane);
  __syncthreads();
  for (int t = lane; t < 256; t += 32) {
    float g = sigmoidf(S->tmp[t]);
#pragma unroll
    for (int d = 0; d < 3; ++d) S->vbuf2[d * 256 + t] *= g;
  }
  __syncthreads();
}

// ---------------- kernels ----------------

// Pre-swizzle one weight (OUT x K, f32 row-major) into f16 B fragments.
__global__ void k_prep(const float* W, int K, int kC, int frags, _Float16* dst) {
  int tid = blockIdx.x * blockDim.x + threadIdx.x;
  int f = tid >> 5, lane = tid & 31;
  if (f >= frags) return;
  int nt = f / kC, kc = f % kC;
  int n = nt * 16 + (lane & 15);
  int kb = kc * 32 + ((lane >> 4) << 3);
  _Float16* o = dst + f * 512 + lane * 16;
  for (int j = 0; j < 16; ++j) {
    int k = kb + (j & 7) + ((j >> 3) << 4);
    o[j] = (k < K) ? (_Float16)W[n * K + k] : (_Float16)0.0f;
  }
}

__global__ void k_init(float* rowmax, float* rowsum, float* agg_s, float* agg_v) {
  int t = blockIdx.x * blockDim.x + threadIdx.x;
  if (t < NN * 32) { rowmax[t] = -3.0e38f; rowsum[t] = 0.0f; }
  if (t < NN * 64) agg_s[t] = 0.0f;
  if (t < NN * 48) agg_v[t] = 0.0f;
}

__global__ void __launch_bounds__(64)
k_nodes(const float* node_sca, const float* node_vec,
        const _Float16* nm_Wv1, const _Float16* nm_Wv2, const _Float16* nm_Ws,
        const _Float16* nm_Wg, const float* nm_bg,
        const _Float16* rs_Wv1, const _Float16* rs_Wv2, const _Float16* rs_Ws,
        const _Float16* rs_Wg, const float* rs_bg,
        const _Float16* Wn, const float* bn, const _Float16* Wns,
        const float* bns, const _Float16* Wnv, const _Float16* Wva,
        float* t_nsca, float* t_nvec, float* t_WnN, float* t_WnsN,
        float* t_WnvN, float* t_hva, float* t_rs, float* t_rv) {
  __shared__ ShmG shm[2];
  int wave = threadIdx.x >> 5, lane = threadIdx.x & 31;
  ShmG* S = &shm[wave];
  int n0 = (blockIdx.x * 2 + wave) * 16;

  for (int t = lane; t < 768; t += 32) {
    int d = t >> 8, r = t & 255, m = r >> 4, i = r & 15;
    S->vin[t] = node_vec[(n0 + m) * 48 + i * 3 + d];
  }
  __syncthreads();
  gv_forward<64, 16, 16, 16>(S, node_sca + n0 * 64, S->vin,
                             nm_Wv1, nm_Wv2, nm_Ws, nm_Wg, nm_bg, lane);
  for (int t = lane; t < 16 * 64; t += 32) t_nsca[n0 * 64 + t] = S->res[t];
  for (int t = lane; t < 768; t += 32) {
    int d = t >> 8, r = t & 255, m = r >> 4, i = r & 15;
    t_nvec[(n0 + m) * 48 + i * 3 + d] = S->vbuf2[t];
  }
  // node projections from nsca
  stage_f32<64, 64, 64>(S->stA, S->res, lane);
  __syncthreads();
  wave_gemm<64, 32, true>(S->stA, 64, Wn, bn, S->tmp2, 32, lane);
  __syncthreads();
  for (int t = lane; t < 16 * 32; t += 32) t_WnN[n0 * 32 + t] = S->tmp2[t];
  wave_gemm<64, 32, true>(S->stA, 64, Wns, bns, S->tmp2, 32, lane);
  __syncthreads();
  for (int t = lane; t < 16 * 32; t += 32) t_WnsN[n0 * 32 + t] = S->tmp2[t];
  // vector projections from nvec
  for (int d = 0; d < 3; ++d) {
    stage_f32<16, 16, 32>(S->stA, S->vbuf2 + d * 256, lane);
    __syncthreads();
    wave_gemm<16, 32, false>(S->stA, 32, Wnv, nullptr, S->tmp2, 32, lane);
    __syncthreads();
    for (int t = lane; t < 16 * 32; t += 32) {
      int m = t >> 5, k = t & 31;
      t_WnvN[(n0 + m) * 96 + k * 3 + d] = S->tmp2[t];
    }
    wave_gemm<16, 32, false>(S->stA, 32, Wva, nullptr, S->tmp2, 32, lane);
    __syncthreads();
    for (int t = lane; t < 16 * 32; t += 32) {
      int m = t >> 5, k = t & 31;
      t_hva[(n0 + m) * 96 + k * 3 + d] = S->tmp2[t];
    }
  }
  // residual branch
  gv_forward<64, 16, 16, 16>(S, node_sca + n0 * 64, S->vin,
                             rs_Wv1, rs_Wv2, rs_Ws, rs_Wg, rs_bg, lane);
  for (int t = lane; t < 16 * 64; t += 32) t_rs[n0 * 64 + t] = S->res[t];
  for (int t = lane; t < 768; t += 32) {
    int d = t >> 8, r = t & 255, m = r >> 4, i = r & 15;
    t_rv[(n0 + m) * 48 + i * 3 + d] = S->vbuf2[t];
  }
}

struct __align__(16) ShmA {
  _Float16 stA[16 * 160];
  float res[16 * 32];
  float dist[16];
  float part[32];
  int rr[16], rc[16], r[16];
};

__global__ void __launch_bounds__(64)
k_edgeA(const float* node_pos, const int* edge_index,
        const float* t_nsca, const float* t_hva,
        const _Float16* Wsa, const float* bsa,
        float* logits, float* av, float* rowmax) {
  __shared__ ShmA shm[2];
  int wave = threadIdx.x >> 5, lane = threadIdx.x & 31;
  ShmA* S = &shm[wave];
  int e0 = (blockIdx.x * 2 + wave) * 16;
  const int* row = edge_index;
  const int* col = edge_index + NE;
  if (lane < 16) {
    int e = e0 + lane;
    int r = row[e], c = col[e];
    S->r[lane] = r;
    S->rr[lane] = row[r];
    S->rc[lane] = row[c];
    float dx = node_pos[r * 3 + 0] - node_pos[c * 3 + 0];
    float dy = node_pos[r * 3 + 1] - node_pos[c * 3 + 1];
    float dz = node_pos[r * 3 + 2] - node_pos[c * 3 + 2];
    S->dist[lane] = sqrtf(dx * dx + dy * dy + dz * dz);
  }
  __syncthreads();
  // attention logits: Wsa @ [nsca[rr], nsca[rc], dist]  (K=129 -> pad 160)
#pragma unroll
  for (int t = lane; t < 16 * 32; t += 32) {  // cols 0..63 (pairs)
    int m = t >> 5, p = t & 31;
    const float* sp = t_nsca + S->rr[m] * 64 + 2 * p;
    v2h h; h[0] = (_Float16)sp[0]; h[1] = (_Float16)sp[1];
    ((v2h*)S->stA)[m * 80 + p] = h;
  }
#pragma unroll
  for (int t = lane; t < 16 * 32; t += 32) {  // cols 64..127 (pairs)
    int m = t >> 5, p = t & 31;
    const float* sp = t_nsca + S->rc[m] * 64 + 2 * p;
    v2h h; h[0] = (_Float16)sp[0]; h[1] = (_Float16)sp[1];
    ((v2h*)S->stA)[m * 80 + 32 + p] = h;
  }
#pragma unroll
  for (int t = lane; t < 16 * 16; t += 32) {  // cols 128..159 (pairs)
    int m = t >> 4, p = t & 15;
    v2h h = {};
    if (p == 0) h[0] = (_Float16)S->dist[m];
    ((v2h*)S->stA)[m * 80 + 64 + p] = h;
  }
  __syncthreads();
  wave_gemm<129, 32, true>(S->stA, 160, Wsa, bsa, S->res, 32, lane);
  __syncthreads();
  for (int t = lane; t < 16 * 32; t += 32) {
    int m = t >> 5, k = t & 31;
    float v = S->res[t];
    logits[(e0 + m) * 32 + k] = v;
    atomicMaxF(rowmax + S->r[m] * 32 + k, v);
  }
  // vector attention a_v = sigmoid(<hva[rr], hva[rc]>)
  {
    int m = lane & 15, h = lane >> 4;
    const float* ha = t_hva + S->rr[m] * 96 + h * 48;
    const float* hb = t_hva + S->rc[m] * 96 + h * 48;
    float s = 0.0f;
#pragma unroll
    for (int j = 0; j < 48; ++j) s += ha[j] * hb[j];
    S->part[lane] = s;
  }
  __syncthreads();
  if (lane < 16) {
    float s = S->part[lane] + S->part[lane + 16];
    av[e0 + lane] = sigmoidf(s);
  }
}

__global__ void k_edgeB(const int* edge_index, const float* rowmax,
                        float* logits, float* rowsum) {
  int t = blockIdx.x * blockDim.x + threadIdx.x;
  if (t >= NE * 32) return;
  int e = t >> 5, k = t & 31;
  int r = edge_index[e];
  float ex = expf(logits[t] - rowmax[r * 32 + k]);
  logits[t] = ex;
  atomicAdd(rowsum + r * 32 + k, ex);
}

__global__ void __launch_bounds__(64)
k_edgeC(const float* node_pos, const int* edge_index, const float* edge_feature,
        const float* W_ee,
        const _Float16* em_Wv1, const _Float16* em_Wv2, const _Float16* em_Ws,
        const _Float16* em_Wg, const float* em_bg,
        const _Float16* We, const float* be, const _Float16* Wes,
        const float* bes, const _Float16* Wev,
        const _Float16* mo_Wv1, const _Float16* mo_Wv2, const _Float16* mo_Ws,
        const _Float16* mo_Wg, const float* mo_bg,
        const float* t_WnN, const float* t_WnsN, const float* t_WnvN,
        const float* logits, const float* av, const float* rowsum,
        float* agg_s, float* agg_v) {
  __shared__ ShmG shm[2];
  int wave = threadIdx.x >> 5, lane = threadIdx.x & 31;
  ShmG* S = &shm[wave];
  int e0 = (blockIdx.x * 2 + wave) * 16;
  const int* row = edge_index;
  const int* col = edge_index + NE;
  if (lane < 16) {
    int e = e0 + lane;
    int r = row[e], c = col[e];
    S->idx[lane] = r;
    S->idx[16 + lane] = row[r];
    float dx = node_pos[r * 3 + 0] - node_pos[c * 3 + 0];
    float dy = node_pos[r * 3 + 1] - node_pos[c * 3 + 1];
    float dz = node_pos[r * 3 + 2] - node_pos[c * 3 + 2];
    float dist = sqrtf(dx * dx + dy * dy + dz * dz);
    float inv = 1.0f / (dist + 1e-7f);
    S->sc[lane] = dist;
    S->sc[16 + lane] = dx * inv;
    S->sc[32 + lane] = dy * inv;
    S->sc[48 + lane] = dz * inv;
    S->sc[64 + lane] = av[e];
  }
  __syncthreads();
  // edge_sca = [12 gaussians, 4 features] -> tmp2 (16x16)
  for (int t = lane; t < 256; t += 32) {
    int m = t >> 4, k = t & 15;
    float v;
    if (k < 12) {
      const float step = 10.0f / 11.0f;
      const float coeff = -0.5f / (step * step);
      float diff = S->sc[m] - step * (float)k;
      v = expf(coeff * diff * diff);
    } else {
      v = edge_feature[(e0 + m) * 4 + (k - 12)];
    }
    S->tmp2[t] = v;
  }
  // edge_vec[d][m][i] = evn[m][d] * W_ee[i] -> vin
  for (int t = lane; t < 768; t += 32) {
    int d = t >> 8, r2 = t & 255, m = r2 >> 4, i = r2 & 15;
    S->vin[t] = S->sc[16 + d * 16 + m] * W_ee[i];
  }
  __syncthreads();
  gv_forward<16, 16, 16, 16>(S, S->tmp2, S->vin,
                             em_Wv1, em_Wv2, em_Ws, em_Wg, em_bg, lane);
  // res = esca (16x64), vbuf2 = evec [d][m][i]
  stage_f32<64, 64, 64>(S->stA, S->res, lane);
  __syncthreads();
  wave_gemm<64, 32, true>(S->stA, 64, We, be, S->tmp2, 32, lane);   // We@esca+be
  wave_gemm<64, 32, true>(S->stA, 64, Wes, bes, S->tmp, 32, lane);  // es_h
  __syncthreads();
  for (int d = 0; d < 3; ++d) {  // ev_h -> vbuf (stride 32)
    stage_f32<16, 16, 32>(S->stA, S->vbuf2 + d * 256, lane);
    __syncthreads();
    wave_gemm<16, 32, false>(S->stA, 32, Wev, nullptr, S->vbuf + d * 512, 32,
                             lane);
    __syncthreads();
  }
  // build msg_s (tmp2) and msg_v (vbuf, in place)
  for (int t = lane; t < 16 * 32; t += 32) {
    int m = t >> 5, k = t & 31;
    int r = S->idx[m], rr = S->idx[16 + m];
    float aS = logits[(e0 + m) * 32 + k] / rowsum[r * 32 + k];
    float nsf = t_WnN[rr * 32 + k] * S->tmp2[t];
    float avv = S->sc[64 + m];
    float nsh = t_WnsN[rr * 32 + k];
    float esh = S->tmp[t];
#pragma unroll
    for (int d = 0; d < 3; ++d) {
      float evh = S->vbuf[d * 512 + t];
      float nvh = t_WnvN[rr * 96 + k * 3 + d];
      S->vbuf[d * 512 + t] = (nsh * evh + nvh * esh) * avv;
    }
    S->tmp2[t] = nsf * aS;
  }
  __syncthreads();
  gv_forward<32, 32, 32, 32>(S, S->tmp2, S->vbuf,
                             mo_Wv1, mo_Wv2, mo_Ws, mo_Wg, mo_bg, lane);
  // aggregate per destination node (segment over row)
  for (int t = lane; t < 16 * 64; t += 32) {
    int m = t >> 6, o = t & 63;
    atomicAdd(agg_s + S->idx[m] * 64 + o, S->res[t]);
  }
  for (int t = lane; t < 768; t += 32) {
    int d = t >> 8, r2 = t & 255, m = r2 >> 4, i = r2 & 15;
    atomicAdd(agg_v + S->idx[m] * 48 + i * 3 + d, S->vbuf2[t]);
  }
}

__global__ void __launch_bounds__(64)
k_final(const float* t_rs, const float* t_rv, const float* agg_s,
        const float* agg_v,
        const _Float16* ag_Wv1, const _Float16* ag_Wv2, const _Float16* ag_Ws,
        const _Float16* ag_Wg, const float* ag_bg,
        const _Float16* Wd, float* out) {
  __shared__ ShmG shm[2];
  int wave = threadIdx.x >> 5, lane = threadIdx.x & 31;
  ShmG* S = &shm[wave];
  int n0 = (blockIdx.x * 2 + wave) * 16;
  for (int t = lane; t < 16 * 64; t += 32)
    S->vbuf[t] = t_rs[n0 * 64 + t] + agg_s[n0 * 64 + t];
  for (int t = lane; t < 768; t += 32) {
    int d = t >> 8, r2 = t & 255, m = r2 >> 4, i = r2 & 15;
    S->vin[t] = t_rv[(n0 + m) * 48 + i * 3 + d] + agg_v[(n0 + m) * 48 + i * 3 + d];
  }
  __syncthreads();
  gv_forward<64, 16, 16, 16>(S, S->vbuf, S->vin,
                             ag_Wv1, ag_Wv2, ag_Ws, ag_Wg, ag_bg, lane);
  // res = o_s0 (16x64), vbuf2 = o_v0. VNLeakyReLU direction map d = Wd @ v.
  for (int d = 0; d < 3; ++d) {
    stage_f32<16, 16, 32>(S->stA, S->vbuf2 + d * 256, lane);
    __syncthreads();
    wave_gemm<16, 16, false>(S->stA, 32, Wd, nullptr, S->vi + d * 256, 16, lane);
    __syncthreads();
  }
  for (int t = lane; t < 256; t += 32) {
    int m = t >> 4, i = t & 15;
    float x0 = S->vbuf2[t], x1 = S->vbuf2[256 + t], x2 = S->vbuf2[512 + t];
    float d0 = S->vi[t],    d1 = S->vi[256 + t],   d2 = S->vi[512 + t];
    float dot = x0 * d0 + x1 * d1 + x2 * d2;
    float dsq = d0 * d0 + d1 * d1 + d2 * d2;
    float tt = dot / (dsq + 1e-6f);
    bool pos = dot >= 0.0f;
    float o0 = 0.2f * x0 + 0.8f * (pos ? x0 : (x0 - tt * d0));
    float o1 = 0.2f * x1 + 0.8f * (pos ? x1 : (x1 - tt * d1));
    float o2 = 0.2f * x2 + 0.8f * (pos ? x2 : (x2 - tt * d2));
    int base = 64 * NN + (n0 + m) * 48 + i * 3;
    out[base + 0] = o0; out[base + 1] = o1; out[base + 2] = o2;
  }
  for (int t = lane; t < 16 * 64; t += 32) {
    float v = S->res[t];
    out[n0 * 64 + t] = (v >= 0.0f) ? v : 0.01f * v;
  }
}

// ---------------- host launcher ----------------

extern "C" void kernel_launch(void* const* d_in, const int* in_sizes, int n_in,
                              void* d_out, int out_size, void* d_ws, size_t ws_size,
                              hipStream_t stream) {
  auto F = [&](int i) { return (const float*)d_in[i]; };
  // top-level dict in insertion order; params pytree leaves sorted by key.
  const float* node_sca = F(0);
  const float* node_vec = F(1);
  const float* node_pos = F(2);
  const float* edge_feature = F(3);
  const int*   edge_index = (const int*)d_in[4];
  const float* W_ee = F(5);
  const float* We = F(6);  const float* Wes = F(7);  const float* Wev = F(8);
  const float* Wn = F(9);  const float* Wns = F(10); const float* Wnv = F(11);
  const float* Wsa = F(12); const float* Wva = F(13);
  const float* ag_Wg = F(14), *ag_Ws = F(15), *ag_Wv1 = F(16), *ag_Wv2 = F(17), *ag_bg = F(18);
  const float* Wd = F(19);
  const float* be = F(20), *bes = F(21), *bn = F(22), *bns = F(23), *bsa = F(24);
  const float* em_Wg = F(25), *em_Ws = F(26), *em_Wv1 = F(27), *em_Wv2 = F(28), *em_bg = F(29);
  const float* mo_Wg = F(30), *mo_Ws = F(31), *mo_Wv1 = F(32), *mo_Wv2 = F(33), *mo_bg = F(34);
  const float* nm_Wg = F(35), *nm_Ws = F(36), *nm_Wv1 = F(37), *nm_Wv2 = F(38), *nm_bg = F(39);
  const float* rs_Wg = F(40), *rs_Ws = F(41), *rs_Wv1 = F(42), *rs_Wv2 = F(43), *rs_bg = F(44);

  float* ws = (float*)d_ws;
  float* t_nsca = ws;                 // 64N
  float* t_nvec = t_nsca + 64 * NN;   // 48N
  float* t_WnN  = t_nvec + 48 * NN;   // 32N
  float* t_WnsN = t_WnN  + 32 * NN;   // 32N
  float* t_WnvN = t_WnsN + 32 * NN;   // 96N
  float* t_hva  = t_WnvN + 96 * NN;   // 96N
  float* t_rs   = t_hva  + 96 * NN;   // 64N
  float* t_rv   = t_rs   + 64 * NN;   // 48N
  float* rowmax = t_rv   + 48 * NN;   // 32N
  float* rowsum = rowmax + 32 * NN;   // 32N
  float* agg_s  = rowsum + 32 * NN;   // 64N
  float* agg_v  = agg_s  + 64 * NN;   // 48N
  float* logits = agg_v  + 48 * NN;   // 32E
  float* av     = logits + 32 * (size_t)NE; // E

  // f16 B-fragment region (32B aligned: float prefix is 94,720,000 bytes)
  _Float16* hbase = (_Float16*)(av + NE);
  size_t hoff = 0;
  auto prep = [&](const float* W, int K, int OUT) -> const _Float16* {
    int kC = (K + 31) / 32, frags = (OUT / 16) * kC;
    _Float16* dst = hbase + hoff;
    k_prep<<<(frags * 32 + 63) / 64, 64, 0, stream>>>(W, K, kC, frags, dst);
    hoff += (size_t)frags * 512;
    return dst;
  };
  const _Float16* Bnm_Wv1 = prep(nm_Wv1, 16, 16);
  const _Float16* Bnm_Wv2 = prep(nm_Wv2, 16, 16);
  const _Float16* Bnm_Ws  = prep(nm_Ws, 80, 64);
  const _Float16* Bnm_Wg  = prep(nm_Wg, 64, 16);
  const _Float16* Brs_Wv1 = prep(rs_Wv1, 16, 16);
  const _Float16* Brs_Wv2 = prep(rs_Wv2, 16, 16);
  const _Float16* Brs_Ws  = prep(rs_Ws, 80, 64);
  const _Float16* Brs_Wg  = prep(rs_Wg, 64, 16);
  const _Float16* BWn     = prep(Wn, 64, 32);
  const _Float16* BWns    = prep(Wns, 64, 32);
  const _Float16* BWnv    = prep(Wnv, 16, 32);
  const _Float16* BWva    = prep(Wva, 16, 32);
  const _Float16* BWsa    = prep(Wsa, 129, 32);
  const _Float16* Bem_Wv1 = prep(em_Wv1, 16, 16);
  const _Float16* Bem_Wv2 = prep(em_Wv2, 16, 16);
  const _Float16* Bem_Ws  = prep(em_Ws, 32, 64);
  const _Float16* Bem_Wg  = prep(em_Wg, 64, 16);
  const _Float16* BWe     = prep(We, 64, 32);
  const _Float16* BWes    = prep(Wes, 64, 32);
  const _Float16* BWev    = prep(Wev, 16, 32);
  const _Float16* Bmo_Wv1 = prep(mo_Wv1, 32, 32);
  const _Float16* Bmo_Wv2 = prep(mo_Wv2, 32, 16);
  const _Float16* Bmo_Ws  = prep(mo_Ws, 64, 64);
  const _Float16* Bmo_Wg  = prep(mo_Wg, 64, 16);
  const _Float16* Bag_Wv1 = prep(ag_Wv1, 16, 16);
  const _Float16* Bag_Wv2 = prep(ag_Wv2, 16, 16);
  const _Float16* Bag_Ws  = prep(ag_Ws, 80, 64);
  const _Float16* Bag_Wg  = prep(ag_Wg, 64, 16);
  const _Float16* BWd     = prep(Wd, 16, 16);

  k_init<<<(NN * 64 + 255) / 256, 256, 0, stream>>>(rowmax, rowsum, agg_s, agg_v);

  k_nodes<<<NN / 32, 64, 0, stream>>>(
      node_sca, node_vec,
      Bnm_Wv1, Bnm_Wv2, Bnm_Ws, Bnm_Wg, nm_bg,
      Brs_Wv1, Brs_Wv2, Brs_Ws, Brs_Wg, rs_bg,
      BWn, bn, BWns, bns, BWnv, BWva,
      t_nsca, t_nvec, t_WnN, t_WnsN, t_WnvN, t_hva, t_rs, t_rv);

  k_edgeA<<<NE / 32, 64, 0, stream>>>(
      node_pos, edge_index, t_nsca, t_hva, BWsa, bsa, logits, av, rowmax);

  k_edgeB<<<(NE * 32 + 255) / 256, 256, 0, stream>>>(edge_index, rowmax, logits, rowsum);

  k_edgeC<<<NE / 32, 64, 0, stream>>>(
      node_pos, edge_index, edge_feature, W_ee,
      Bem_Wv1, Bem_Wv2, Bem_Ws, Bem_Wg, em_bg,
      BWe, be, BWes, bes, BWev,
      Bmo_Wv1, Bmo_Wv2, Bmo_Ws, Bmo_Wg, mo_bg,
      t_WnN, t_WnsN, t_WnvN,
      logits, av, rowsum, agg_s, agg_v);

  k_final<<<NN / 32, 64, 0, stream>>>(
      t_rs, t_rv, agg_s, agg_v,
      Bag_Wv1, Bag_Wv2, Bag_Ws, Bag_Wg, ag_bg, BWd, (float*)d_out);

  (void)in_sizes; (void)n_in; (void)out_size; (void)ws_size;
}
